// MultiscaleMambaBottleneck_58342835749264
// MI455X (gfx1250) — compile-verified, matrix operand order checked by reading, not run on hardware
//
#include <hip/hip_runtime.h>

// ---------------------------------------------------------------------------
// MultiscaleMambaBottleneck forward for MI455X (gfx1250, wave32, WMMA).
// GEMMs use v_wmma_f32_16x16x32_f16 (f16 in, f32 acc), register-blocked
// 16x64 per wave where N % 64 == 0. The sequential selective scan runs in
// VALU with register-resident state.
// ---------------------------------------------------------------------------

typedef __attribute__((ext_vector_type(8)))  _Float16 v8h;
typedef __attribute__((ext_vector_type(16))) _Float16 v16h;
typedef __attribute__((ext_vector_type(8)))  float    v8f;

#define NB_B     4
#define NB_L     2304      // 48*48
#define NB_BL    9216      // 4*2304
#define NB_DIM   192
#define NB_DIN   128

__device__ __forceinline__ v16h cat8(v8h lo, v8h hi) {
  return __builtin_shufflevector(lo, hi, 0,1,2,3,4,5,6,7,8,9,10,11,12,13,14,15);
}

// ---------------------------------------------------------------------------
// fp32 -> fp16 convert
__global__ void k_cvt(const float* __restrict__ s, _Float16* __restrict__ d, int n) {
  int t = blockIdx.x * blockDim.x + threadIdx.x;
  if (t < n) d[t] = (_Float16)s[t];
}

// x (B, 192, HW) fp32 -> xh (B*HW, 192) fp16  (channels-last for GEMM A)
__global__ void k_xpose(const float* __restrict__ x, _Float16* __restrict__ xh) {
  int t = blockIdx.x * blockDim.x + threadIdx.x;
  if (t >= NB_BL * NB_DIM) return;
  int c  = t % NB_DIM;
  int bl = t / NB_DIM;
  int l  = bl % NB_L;
  int b  = bl / NB_L;
  xh[t] = (_Float16)x[((size_t)(b * NB_DIM + c)) * NB_L + l];
}

// c2_w (O,I,3,3) fp32 -> w2t (O, (kh*3+kw)*192 + c) fp16 for implicit GEMM
__global__ void k_w2t(const float* __restrict__ w2, _Float16* __restrict__ w2t) {
  int t = blockIdx.x * blockDim.x + threadIdx.x;
  if (t >= NB_DIM * 1728) return;
  int o  = t / 1728;
  int r  = t % 1728;
  int kk = r / NB_DIM;
  int c  = r % NB_DIM;
  w2t[t] = (_Float16)w2[((size_t)(o * NB_DIM) + c) * 9 + kk];
}

// ---------------------------------------------------------------------------
// LayerNorm over 64 channels (one wave32 per row), writes f16 GEMM-A operand.
__global__ void k_layernorm(const float* __restrict__ seq, int col0,
                            const float* __restrict__ w, const float* __restrict__ b,
                            _Float16* __restrict__ out) {
  int wid  = (blockIdx.x * blockDim.x + threadIdx.x) >> 5;
  int lane = threadIdx.x & 31;
  if (wid >= NB_BL) return;
  const float* rp = seq + (size_t)wid * NB_DIM + col0;
  float v0 = rp[lane], v1 = rp[lane + 32];
  float s = v0 + v1, q = v0 * v0 + v1 * v1;
#pragma unroll
  for (int m = 16; m >= 1; m >>= 1) {
    s += __shfl_xor(s, m, 32);
    q += __shfl_xor(q, m, 32);
  }
  float mean = s * (1.f / 64.f);
  float var  = q * (1.f / 64.f) - mean * mean;
  float inv  = rsqrtf(var + 1e-5f);
  out[(size_t)wid * 64 + lane]      = (_Float16)((v0 - mean) * inv * w[lane]      + b[lane]);
  out[(size_t)wid * 64 + lane + 32] = (_Float16)((v1 - mean) * inv * w[lane + 32] + b[lane + 32]);
}

// ---------------------------------------------------------------------------
// Generic WMMA GEMM (16x16 tile per wave) for odd N (x_proj: N=36/68/100).
// C[M,N] (+bias, optional +=C) = A[M,K] * W[N,K]^T
__global__ void k_gemm(const _Float16* __restrict__ A, int lda,
                       const _Float16* __restrict__ W, int ldw,
                       float* __restrict__ C, int ldc, int col0,
                       int N, int K,
                       const float* __restrict__ bias, int addC) {
  int wid  = (blockIdx.x * blockDim.x + threadIdx.x) >> 5;
  int lane = threadIdx.x & 31;
  int Nt = (N + 15) >> 4;
  int tiles = (NB_BL >> 4) * Nt;
  if (wid >= tiles) return;
  int m0 = (wid / Nt) << 4;
  int n0 = (wid % Nt) << 4;
  int r  = lane & 15;
  int ko = (lane >> 4) << 3;                 // 0 or 8 (K sub-offset / C row offset)
  const _Float16* ap = A + (size_t)(m0 + r) * lda + ko;
  int wr = n0 + r;
  const _Float16* wp = W + (size_t)(wr < N ? wr : 0) * ldw + ko;
  v8f acc = {};
  for (int k0 = 0; k0 < K; k0 += 32) {
    v16h af = cat8(*(const v8h*)(ap + k0), *(const v8h*)(ap + k0 + 16));
    v16h wf = cat8(*(const v8h*)(wp + k0), *(const v8h*)(wp + k0 + 16));
    acc = __builtin_amdgcn_wmma_f32_16x16x32_f16(false, af, false, wf,
                                                 (short)0, acc, false, false);
  }
  int col = n0 + r;
  if (col < N) {
    float badd = bias ? bias[col] : 0.f;
    int rbase = m0 + ko;
#pragma unroll
    for (int v = 0; v < 8; ++v) {
      float* cp = C + (size_t)(rbase + v) * ldc + col0 + col;
      float val = acc[v] + badd;
      if (addC) val += *cp;
      *cp = val;
    }
  }
}

// ---------------------------------------------------------------------------
// Register-blocked WMMA GEMM: one wave computes a 16x64 tile (4 WMMAs/K-step,
// A fragment loaded once and reused). Requires N % 64 == 0.
__global__ void k_gemm4(const _Float16* __restrict__ A, int lda,
                        const _Float16* __restrict__ W, int ldw,
                        float* __restrict__ C, int ldc, int col0,
                        int N, int K,
                        const float* __restrict__ bias, int addC) {
  int wid  = (blockIdx.x * blockDim.x + threadIdx.x) >> 5;
  int lane = threadIdx.x & 31;
  int Nt = N >> 6;
  int tiles = (NB_BL >> 4) * Nt;
  if (wid >= tiles) return;
  int m0 = (wid / Nt) << 4;
  int n0 = (wid % Nt) << 6;
  int r  = lane & 15;
  int ko = (lane >> 4) << 3;
  const _Float16* ap = A + (size_t)(m0 + r) * lda + ko;
  const _Float16* wp = W + (size_t)(n0 + r) * ldw + ko;
  size_t wstep = (size_t)16 * ldw;
  v8f acc[4] = {};
  for (int k0 = 0; k0 < K; k0 += 32) {
    v16h af = cat8(*(const v8h*)(ap + k0), *(const v8h*)(ap + k0 + 16));
#pragma unroll
    for (int j = 0; j < 4; ++j) {
      const _Float16* w0 = wp + (size_t)j * wstep;
      v16h wf = cat8(*(const v8h*)(w0 + k0), *(const v8h*)(w0 + k0 + 16));
      acc[j] = __builtin_amdgcn_wmma_f32_16x16x32_f16(false, af, false, wf,
                                                      (short)0, acc[j], false, false);
    }
  }
  int rbase = m0 + ko;
#pragma unroll
  for (int j = 0; j < 4; ++j) {
    int col = n0 + 16 * j + r;
    float badd = bias ? bias[col] : 0.f;
#pragma unroll
    for (int v = 0; v < 8; ++v) {
      float* cp = C + (size_t)(rbase + v) * ldc + col0 + col;
      float val = acc[j][v] + badd;
      if (addC) val += *cp;
      *cp = val;
    }
  }
}

// ---------------------------------------------------------------------------
// Causal depthwise conv1d (width 4) + SiLU on xi = xz[:, :128]
__global__ void k_conv_silu(const float* __restrict__ xz,
                            const float* __restrict__ cw, const float* __restrict__ cb,
                            float* __restrict__ xc, _Float16* __restrict__ xch) {
  int t = blockIdx.x * blockDim.x + threadIdx.x;
  if (t >= NB_BL * NB_DIN) return;
  int d  = t & 127;
  int bl = t >> 7;
  int l  = bl % NB_L;
  int b  = bl / NB_L;
  float acc = cb[d];
#pragma unroll
  for (int k = 0; k < 4; ++k) {
    int ll = l - 3 + k;
    if (ll >= 0) acc += xz[(((size_t)(b * NB_L + ll)) << 8) + d] * cw[d * 4 + k];
  }
  float sv = acc / (1.f + expf(-acc));
  xc[t]  = sv;
  xch[t] = (_Float16)sv;
}

// delta = softplus(dt @ dt_w^T + dt_b), dt = dbl[:, 0:4]
__global__ void k_delta(const float* __restrict__ dbl, int ldd,
                        const float* __restrict__ dtw, const float* __restrict__ dtb,
                        float* __restrict__ delta) {
  int t = blockIdx.x * blockDim.x + threadIdx.x;
  if (t >= NB_BL * NB_DIN) return;
  int d  = t & 127;
  int bl = t >> 7;
  const float* dp = dbl + (size_t)bl * ldd;
  float s = dtb[d];
#pragma unroll
  for (int r = 0; r < 4; ++r) s += dp[r] * dtw[d * 4 + r];
  delta[t] = (s > 20.f) ? s : log1pf(expf(s));
}

// ---------------------------------------------------------------------------
// Selective scan (sequential over L). One block per batch b, 128 threads = d_in.
// State h[S] and -exp(A_log) live in registers (S is a template constant).
// Writes gated output (y * silu(z)) as f16 -> A operand of out_proj GEMM.
template <int S>
__global__ void k_scan(const float* __restrict__ delta,
                       const float* __restrict__ xc,
                       const float* __restrict__ dbl, int ldd,
                       const float* __restrict__ Alog,
                       const float* __restrict__ Dp,
                       const float* __restrict__ xz,
                       _Float16* __restrict__ yg) {
  int b = blockIdx.x;
  int d = threadIdx.x;  // 0..127
  float Anz[S], hst[S];
#pragma unroll
  for (int s = 0; s < S; ++s) {
    Anz[s] = -expf(Alog[d * S + s]);
    hst[s] = 0.f;
  }
  float Dv = Dp[d];
  __shared__ float sB[S], sC[S];
  for (int l = 0; l < NB_L; ++l) {
    int bl = b * NB_L + l;
    if (d < S)             sB[d]     = dbl[(size_t)bl * ldd + 4 + d];
    else if (d < 2 * S)    sC[d - S] = dbl[(size_t)bl * ldd + 4 + S + (d - S)];
    __syncthreads();
    float dlt = delta[((size_t)bl << 7) + d];
    float xv  = xc[((size_t)bl << 7) + d];
    float dx  = dlt * xv;
    float acc = 0.f;
#pragma unroll
    for (int s = 0; s < S; ++s) {
      float hs = expf(dlt * Anz[s]) * hst[s] + dx * sB[s];
      hst[s] = hs;
      acc += hs * sC[s];
    }
    float zv   = xz[(((size_t)bl) << 8) + 128 + d];
    float gate = zv / (1.f + expf(-zv));
    yg[((size_t)bl << 7) + d] = (_Float16)((acc + Dv * xv) * gate);
    __syncthreads();
  }
}

// ---------------------------------------------------------------------------
// Fuse 1x1 conv: blocked WMMA GEMM (16x64/wave) + bias + BN + ReLU -> f16.
__global__ void k_gemm_c1(const _Float16* __restrict__ A,
                          const _Float16* __restrict__ W,
                          const float* __restrict__ bias,
                          const float* __restrict__ g, const float* __restrict__ bb,
                          const float* __restrict__ mm, const float* __restrict__ vv,
                          _Float16* __restrict__ out) {
  int wid  = (blockIdx.x * blockDim.x + threadIdx.x) >> 5;
  int lane = threadIdx.x & 31;
  const int Nt = NB_DIM / 64;                // 3
  if (wid >= (NB_BL >> 4) * Nt) return;
  int m0 = (wid / Nt) << 4;
  int n0 = (wid % Nt) << 6;
  int r  = lane & 15;
  int ko = (lane >> 4) << 3;
  const _Float16* ap = A + (size_t)(m0 + r) * NB_DIM + ko;
  const _Float16* wp = W + (size_t)(n0 + r) * NB_DIM + ko;
  const size_t wstep = (size_t)16 * NB_DIM;
  v8f acc[4] = {};
  for (int k0 = 0; k0 < NB_DIM; k0 += 32) {
    v16h af = cat8(*(const v8h*)(ap + k0), *(const v8h*)(ap + k0 + 16));
#pragma unroll
    for (int j = 0; j < 4; ++j) {
      const _Float16* w0 = wp + (size_t)j * wstep;
      v16h wf = cat8(*(const v8h*)(w0 + k0), *(const v8h*)(w0 + k0 + 16));
      acc[j] = __builtin_amdgcn_wmma_f32_16x16x32_f16(false, af, false, wf,
                                                      (short)0, acc[j], false, false);
    }
  }
  int rbase = m0 + ko;
#pragma unroll
  for (int j = 0; j < 4; ++j) {
    int col = n0 + 16 * j + r;
    float inv = g[col] * rsqrtf(vv[col] + 1e-5f);
    float sh  = bb[col] - mm[col] * inv;
    float bd  = bias[col];
#pragma unroll
    for (int v = 0; v < 8; ++v) {
      float t = (acc[j][v] + bd) * inv + sh;
      out[(size_t)(rbase + v) * NB_DIM + col] = (_Float16)fmaxf(t, 0.f);
    }
  }
}

// 3x3 conv as implicit blocked WMMA GEMM (K = 9*192), A gathered once per
// K-step and reused across 4 N-tiles; + bias + BN + ReLU -> NCHW fp32.
__device__ __forceinline__ v8h gather9(const _Float16* __restrict__ A,
                                       int b, int h, int w, int kc) {
  int kk = kc / NB_DIM;           // 0..8  -> (kh,kw)
  int c  = kc - kk * NB_DIM;
  int hh = h + (kk / 3) - 1;
  int ww = w + (kk % 3) - 1;
  if ((unsigned)hh < 48u && (unsigned)ww < 48u)
    return *(const v8h*)(A + ((size_t)(b * NB_L + hh * 48 + ww)) * NB_DIM + c);
  v8h z = {};
  return z;
}

__global__ void k_conv3x3(const _Float16* __restrict__ A,   // (BL,192) f16
                          const _Float16* __restrict__ W,   // (192,1728) f16
                          const float* __restrict__ bias,
                          const float* __restrict__ g, const float* __restrict__ bb,
                          const float* __restrict__ mm, const float* __restrict__ vv,
                          float* __restrict__ out) {        // (B,192,HW) f32
  int wid  = (blockIdx.x * blockDim.x + threadIdx.x) >> 5;
  int lane = threadIdx.x & 31;
  const int Nt = NB_DIM / 64;                // 3
  if (wid >= (NB_BL >> 4) * Nt) return;
  int m0 = (wid / Nt) << 4;
  int n0 = (wid % Nt) << 6;
  int r  = lane & 15;
  int ko = (lane >> 4) << 3;
  int mrow = m0 + r;
  int b  = mrow / NB_L;
  int hw = mrow % NB_L;
  int h  = hw / 48, w = hw % 48;
  const _Float16* wp = W + (size_t)(n0 + r) * 1728 + ko;
  const size_t wstep = (size_t)16 * 1728;
  v8f acc[4] = {};
  for (int k0 = 0; k0 < 1728; k0 += 32) {
    v16h af = cat8(gather9(A, b, h, w, k0 + ko), gather9(A, b, h, w, k0 + 16 + ko));
#pragma unroll
    for (int j = 0; j < 4; ++j) {
      const _Float16* w0 = wp + (size_t)j * wstep;
      v16h wf = cat8(*(const v8h*)(w0 + k0), *(const v8h*)(w0 + k0 + 16));
      acc[j] = __builtin_amdgcn_wmma_f32_16x16x32_f16(false, af, false, wf,
                                                      (short)0, acc[j], false, false);
    }
  }
  int rbase = m0 + ko;
#pragma unroll
  for (int j = 0; j < 4; ++j) {
    int col = n0 + 16 * j + r;
    float inv = g[col] * rsqrtf(vv[col] + 1e-5f);
    float sh  = bb[col] - mm[col] * inv;
    float bd  = bias[col];
#pragma unroll
    for (int v = 0; v < 8; ++v) {
      int row = rbase + v;
      int b2  = row / NB_L;
      int hw2 = row % NB_L;
      float t = (acc[j][v] + bd) * inv + sh;
      out[((size_t)(b2 * NB_DIM + col)) * NB_L + hw2] = fmaxf(t, 0.f);
    }
  }
}

// ---------------------------------------------------------------------------
extern "C" void kernel_launch(void* const* d_in, const int* in_sizes, int n_in,
                              void* d_out, int out_size, void* d_ws, size_t ws_size,
                              hipStream_t stream) {
  (void)in_sizes; (void)n_in; (void)out_size; (void)ws_size;
  const int BL = NB_BL;
  const float* x = (const float*)d_in[0];
  auto F = [&](int i) { return (const float*)d_in[i]; };

  // Workspace bump allocator (256B aligned).
  char* base = (char*)d_ws;
  size_t off = 0;
  auto alloc = [&](size_t bytes) -> char* {
    char* p = base + off;
    off += (bytes + 255) & ~(size_t)255;
    return p;
  };
  _Float16* xh   = (_Float16*)alloc((size_t)BL * 192 * 2);
  float*    seqA = (float*)   alloc((size_t)BL * 192 * 4);
  _Float16* hln  = (_Float16*)alloc((size_t)BL * 64 * 2);
  float*    xz   = (float*)   alloc((size_t)BL * 256 * 4);
  float*    xc   = (float*)   alloc((size_t)BL * 128 * 4);
  _Float16* xch  = (_Float16*)alloc((size_t)BL * 128 * 2);
  float*    dbl  = (float*)   alloc((size_t)BL * 100 * 4);
  float*    dlt  = (float*)   alloc((size_t)BL * 128 * 4);
  _Float16* yg   = (_Float16*)alloc((size_t)BL * 128 * 2);
  _Float16* yh   = (_Float16*)alloc((size_t)BL * 192 * 2);
  _Float16* t1h  = (_Float16*)alloc((size_t)BL * 192 * 2);
  _Float16* w2t  = (_Float16*)alloc((size_t)192 * 1728 * 2);
  _Float16* whIn = (_Float16*)alloc(256 * 64 * 2);
  _Float16* whXp = (_Float16*)alloc(100 * 128 * 2);
  _Float16* whOut= (_Float16*)alloc(64 * 128 * 2);
  _Float16* whPr = (_Float16*)alloc(64 * 192 * 2);
  _Float16* whC1 = (_Float16*)alloc(192 * 192 * 2);

  auto cvt = [&](const float* s, _Float16* d, int n) {
    k_cvt<<<(n + 255) / 256, 256, 0, stream>>>(s, d, n);
  };
  auto gemm = [&](const _Float16* A, int lda, const _Float16* W, int ldw,
                  float* C, int ldc, int col0, int N, int K,
                  const float* bias, int addC) {
    if ((N & 63) == 0) {
      int tiles = (BL >> 4) * (N >> 6);
      k_gemm4<<<(tiles + 7) / 8, 256, 0, stream>>>(A, lda, W, ldw, C, ldc, col0,
                                                   N, K, bias, addC);
    } else {
      int tiles = (BL >> 4) * ((N + 15) >> 4);
      k_gemm<<<(tiles + 7) / 8, 256, 0, stream>>>(A, lda, W, ldw, C, ldc, col0,
                                                  N, K, bias, addC);
    }
  };

  // x -> (BL,192) f16 once.
  k_xpose<<<(BL * 192 + 255) / 256, 256, 0, stream>>>(x, xh);

  for (int br = 0; br < 3; ++br) {
    const int pbase = 1 + br * 24;          // proj_w, proj_b, then 2 blocks * 11
    const int S = 16 * (br + 1);
    const int Ndbl = 4 + 2 * S;
    const int col0 = 64 * br;

    // Branch projection: seqA[:, col0:col0+64] = xh @ proj_w^T + proj_b
    cvt(F(pbase + 0), whPr, 64 * 192);
    gemm(xh, 192, whPr, 192, seqA, 192, col0, 64, 192, F(pbase + 1), 0);

    for (int blk = 0; blk < 2; ++blk) {
      const int q = pbase + 2 + blk * 11;   // ln_w,ln_b,in_w,conv_w,conv_b,
                                            // xproj_w,dt_w,dt_b,A_log,D,out_w
      cvt(F(q + 2), whIn, 256 * 64);
      cvt(F(q + 5), whXp, Ndbl * 128);
      cvt(F(q + 10), whOut, 64 * 128);

      // LayerNorm -> hln (f16)
      k_layernorm<<<(BL * 32 + 255) / 256, 256, 0, stream>>>(seqA, col0,
                                                             F(q + 0), F(q + 1), hln);
      // in_proj: xz = hln @ in_w^T   (N=256, K=64)
      gemm(hln, 64, whIn, 64, xz, 256, 0, 256, 64, nullptr, 0);
      // causal conv1d + SiLU
      k_conv_silu<<<(BL * 128 + 255) / 256, 256, 0, stream>>>(xz, F(q + 3), F(q + 4),
                                                              xc, xch);
      // x_proj: dbl = xch @ xproj_w^T  (N=Ndbl, K=128)
      gemm(xch, 128, whXp, 128, dbl, Ndbl, 0, Ndbl, 128, nullptr, 0);
      // delta = softplus(dt @ dt_w^T + dt_b)
      k_delta<<<(BL * 128 + 255) / 256, 256, 0, stream>>>(dbl, Ndbl,
                                                          F(q + 6), F(q + 7), dlt);
      // selective scan + gating -> yg (f16)
      if (S == 16)
        k_scan<16><<<NB_B, 128, 0, stream>>>(dlt, xc, dbl, Ndbl, F(q + 8), F(q + 9), xz, yg);
      else if (S == 32)
        k_scan<32><<<NB_B, 128, 0, stream>>>(dlt, xc, dbl, Ndbl, F(q + 8), F(q + 9), xz, yg);
      else
        k_scan<48><<<NB_B, 128, 0, stream>>>(dlt, xc, dbl, Ndbl, F(q + 8), F(q + 9), xz, yg);
      // out_proj + residual: seqA[:, col0:+64] += yg @ out_w^T
      gemm(yg, 128, whOut, 128, seqA, 192, col0, 64, 128, nullptr, 1);
    }
  }

  // Fuse stage.
  const int fb = 73;  // c1_w,c1_b,bn1{g,b,m,v},c2_w,c2_b,bn2{g,b,m,v}
  cvt(seqA, yh, BL * 192);
  cvt(F(fb + 0), whC1, 192 * 192);
  {
    int tiles = (BL >> 4) * (NB_DIM / 64);
    k_gemm_c1<<<(tiles + 7) / 8, 256, 0, stream>>>(yh, whC1, F(fb + 1),
        F(fb + 2), F(fb + 3), F(fb + 4), F(fb + 5), t1h);
  }
  k_w2t<<<(192 * 1728 + 255) / 256, 256, 0, stream>>>(F(fb + 6), w2t);
  {
    int tiles = (BL >> 4) * (NB_DIM / 64);
    k_conv3x3<<<(tiles + 7) / 8, 256, 0, stream>>>(t1h, w2t, F(fb + 7),
        F(fb + 8), F(fb + 9), F(fb + 10), F(fb + 11), (float*)d_out);
  }
}
